// DiscriminatorLSTM_12317966205497
// MI455X (gfx1250) — matrix operand order repaired and checked by reading, not compiled, not run on hardware
//
#include <hip/hip_runtime.h>
#include <hip/hip_bf16.h>

// ---------------- problem dims ----------------
#define B_      128
#define T_      64
#define FEAT_   4096
#define FUSION_ 1024
#define H_      1024
#define G4_     4096   // 4*H

typedef __bf16 bf16;
typedef __attribute__((ext_vector_type(16))) __bf16 v16bf;
typedef __attribute__((ext_vector_type(8)))  __bf16 v8bf;
typedef __attribute__((ext_vector_type(8)))  float  v8f;

// ---------------------------------------------------------------------------
// WMMA operand loaders (bf16, wave32). Memory is row-major [row][k] with
// row-stride `ld` (elements). Each is 2x global_load_b128 per lane.
//
// A (16x32, MxK), per ISA 7.12.2: lane r<16 -> row r, K={0..7,16..23};
//                                 lane 16+r -> row r, K={8..15,24..31}.
__device__ __forceinline__ v16bf load_a(const bf16* __restrict__ base, long ld, int lane) {
    const int row = lane & 15;
    const int k0  = (lane & 16) ? 8 : 0;
    const bf16* p = base + (long)row * ld + k0;
    v8bf lo = *(const v8bf*)(p);        // K = k0 .. k0+7
    v8bf hi = *(const v8bf*)(p + 16);   // K = k0+16 .. k0+23
    v16bf r;
#pragma unroll
    for (int i = 0; i < 8; ++i) { r[i] = lo[i]; r[i + 8] = hi[i]; }
    return r;
}

// B (32x16, KxN) from weight rows W[n][k]: lane c<16 -> col c, K={0..7,16..23};
//                                          lane 16+c -> col c, K={8..15,24..31}.
__device__ __forceinline__ v16bf load_b(const bf16* __restrict__ base, long ld, int lane) {
    const int col = lane & 15;
    const int k0  = (lane & 16) ? 8 : 0;
    const bf16* p = base + (long)col * ld + k0;
    v8bf lo = *(const v8bf*)(p);        // K = k0 .. k0+7
    v8bf hi = *(const v8bf*)(p + 16);   // K = k0+16 .. k0+23
    v16bf r;
#pragma unroll
    for (int i = 0; i < 8; ++i) { r[i] = lo[i]; r[i + 8] = hi[i]; }
    return r;
}

#define WMMA_BF16(A, Bv, Cv) \
    __builtin_amdgcn_wmma_f32_16x16x32_bf16(false, (A), false, (Bv), (short)0, (Cv), false, false)

__device__ __forceinline__ float sigmoidf_(float x) {
    return 1.0f / (1.0f + __expf(-x));
}
__device__ __forceinline__ float tanhf_(float x) {
    float ax = fabsf(x);
    float e  = __expf(-2.0f * ax);
    float t  = (1.0f - e) / (1.0f + e);
    return copysignf(t, x);
}

// ---------------------------------------------------------------------------
// fp32 -> bf16 conversion (one-time; makes weights + x L2-resident in bf16)
__global__ void cvt_bf16_kernel(const float* __restrict__ src, bf16* __restrict__ dst, long n) {
    long i = (long)blockIdx.x * blockDim.x + threadIdx.x;
    const long stride = (long)gridDim.x * blockDim.x;
    for (; i < n; i += stride) dst[i] = (bf16)src[i];
}

// zero h (bf16) and c (fp32) recurrent state
__global__ void init_state_kernel(bf16* __restrict__ h0, float* __restrict__ c, int n) {
    int i = blockIdx.x * blockDim.x + threadIdx.x;
    if (i < n) { h0[i] = (bf16)0.0f; c[i] = 0.0f; }
}

// ---------------------------------------------------------------------------
// Batched feature GEMM: F = relu(X @ W_feat^T + b_feat)
// M = B*T = 8192, N = FUSION = 1024, K = FEAT = 4096.
// One wave computes a 32x32 output tile (2x2 WMMA blocking: 4 loads / 4 WMMA).
__global__ void feat_gemm_kernel(const bf16* __restrict__ Xb,
                                 const bf16* __restrict__ Wfb,
                                 const float* __restrict__ bias,
                                 bf16* __restrict__ F) {
    const int lane = threadIdx.x & 31;
    const int w    = (int)((blockIdx.x * blockDim.x + threadIdx.x) >> 5);
    const int NT   = FUSION_ / 32;                 // 32 n-tiles
    const long m0  = (long)(w / NT) * 32;
    const long n0  = (long)(w % NT) * 32;

    v8f acc[2][2] = {};
    const bf16* A0 = Xb  + m0 * FEAT_;
    const bf16* A1 = Xb  + (m0 + 16) * FEAT_;
    const bf16* B0 = Wfb + n0 * FEAT_;
    const bf16* B1 = Wfb + (n0 + 16) * FEAT_;

    for (int k = 0; k < FEAT_; k += 32) {
        v16bf a0 = load_a(A0 + k, FEAT_, lane);
        v16bf a1 = load_a(A1 + k, FEAT_, lane);
        v16bf b0 = load_b(B0 + k, FEAT_, lane);
        v16bf b1 = load_b(B1 + k, FEAT_, lane);
        acc[0][0] = WMMA_BF16(a0, b0, acc[0][0]);
        acc[0][1] = WMMA_BF16(a0, b1, acc[0][1]);
        acc[1][0] = WMMA_BF16(a1, b0, acc[1][0]);
        acc[1][1] = WMMA_BF16(a1, b1, acc[1][1]);
    }

    // C/D layout: lane half selects M offset 0/8; VGPR j selects M row j.
    const int col   = lane & 15;
    const int rhalf = (lane >> 4) * 8;
#pragma unroll
    for (int ni = 0; ni < 2; ++ni) {
        const long n = n0 + ni * 16 + col;
        const float bv = bias[n];
#pragma unroll
        for (int mi = 0; mi < 2; ++mi) {
#pragma unroll
            for (int jj = 0; jj < 8; ++jj) {
                float v = acc[mi][ni][jj] + bv;
                v = v > 0.0f ? v : 0.0f;
                F[(m0 + mi * 16 + rhalf + jj) * FUSION_ + n] = (bf16)v;
            }
        }
    }
}

// ---------------------------------------------------------------------------
// Fused recurrent step: gates = F_t @ W_ih^T + h @ W_hh^T + b_ih + b_hh,
// then LSTM cell update, all in one kernel.
// Each wave owns all FOUR gate tiles (i,f,g,o) of a (16-batch x 16-H) tile:
// the A operand (f_t rows / h rows) is reused 4x per K-chunk, and the gate
// coupling (n, n+H, n+2H, n+3H) stays wave-local so the cell epilogue fuses in.
__global__ void lstm_step_kernel(const bf16* __restrict__ F,
                                 const bf16* __restrict__ Wihb,
                                 const bf16* __restrict__ Whhb,
                                 const float* __restrict__ b_ih,
                                 const float* __restrict__ b_hh,
                                 const bf16* __restrict__ h_in,   // [B][H] bf16 (prev step)
                                 bf16* __restrict__ h_out,        // [B][H] bf16 (this step)
                                 float* __restrict__ h_out_f32,   // [B][H] f32 (for classifier)
                                 float* __restrict__ c_state,     // [B][H] f32, in-place
                                 int t) {
    const int lane = threadIdx.x & 31;
    const int w    = (int)((blockIdx.x * blockDim.x + threadIdx.x) >> 5);  // 0..511
    const int mt   = w & 7;     // batch tile (128/16)
    const int j    = w >> 3;    // H column tile (1024/16)
    const long m0  = (long)mt * 16;

    v8f acc[4] = {};            // i, f, g, o tiles

    // ---- part 1: f_t @ W_ih^T   (A rows live at F[(b*T + t)*FUSION]) ----
    {
        const bf16* Abase = F + ((long)m0 * T_ + t) * FUSION_;
        const long  Ald   = (long)T_ * FUSION_;
        for (int k = 0; k < FUSION_; k += 32) {
            v16bf a = load_a(Abase + k, Ald, lane);
#pragma unroll
            for (int g = 0; g < 4; ++g) {
                const bf16* Bb = Wihb + ((long)g * H_ + (long)j * 16) * FUSION_ + k;
                v16bf b = load_b(Bb, FUSION_, lane);
                acc[g] = WMMA_BF16(a, b, acc[g]);
            }
        }
    }
    // ---- part 2: h_{t-1} @ W_hh^T ----
    {
        const bf16* Abase = h_in + m0 * H_;
        for (int k = 0; k < H_; k += 32) {
            v16bf a = load_a(Abase + k, H_, lane);
#pragma unroll
            for (int g = 0; g < 4; ++g) {
                const bf16* Bb = Whhb + ((long)g * H_ + (long)j * 16) * H_ + k;
                v16bf b = load_b(Bb, H_, lane);
                acc[g] = WMMA_BF16(a, b, acc[g]);
            }
        }
    }

    // ---- fused LSTM cell epilogue ----
    const int col   = lane & 15;
    const int n     = j * 16 + col;
    const int rhalf = (lane >> 4) * 8;
    const float bi = b_ih[n]          + b_hh[n];
    const float bf = b_ih[H_ + n]     + b_hh[H_ + n];
    const float bg = b_ih[2 * H_ + n] + b_hh[2 * H_ + n];
    const float bo = b_ih[3 * H_ + n] + b_hh[3 * H_ + n];

#pragma unroll
    for (int jj = 0; jj < 8; ++jj) {
        const long m   = m0 + rhalf + jj;
        const long idx = m * H_ + n;
        const float iv = sigmoidf_(acc[0][jj] + bi);
        const float fv = sigmoidf_(acc[1][jj] + bf);
        const float gv = tanhf_  (acc[2][jj] + bg);
        const float ov = sigmoidf_(acc[3][jj] + bo);
        const float cn = fv * c_state[idx] + iv * gv;   // element owned by this wave only
        c_state[idx]   = cn;
        const float hn = ov * tanhf_(cn);
        h_out_f32[idx] = hn;
        h_out[idx]     = (bf16)hn;
    }
}

// ---------------------------------------------------------------------------
// Tiny classifier: s[b, t, 0:2] = h_t[b] @ W_cls^T + b_cls. One wave per row.
__global__ void classify_kernel(const float* __restrict__ h,
                                const float* __restrict__ W_cls,
                                const float* __restrict__ b_cls,
                                float* __restrict__ out, int t) {
    const int lane = threadIdx.x & 31;
    const int b    = (int)((blockIdx.x * blockDim.x + threadIdx.x) >> 5);  // 0..127
    const float* hr = h + (long)b * H_;
    float s0 = 0.0f, s1 = 0.0f;
    for (int k = lane; k < H_; k += 32) {
        const float hv = hr[k];
        s0 += hv * W_cls[k];
        s1 += hv * W_cls[H_ + k];
    }
#pragma unroll
    for (int off = 16; off > 0; off >>= 1) {
        s0 += __shfl_down(s0, off, 32);
        s1 += __shfl_down(s1, off, 32);
    }
    if (lane == 0) {
        const long o = ((long)b * T_ + t) * 2;
        out[o]     = s0 + b_cls[0];
        out[o + 1] = s1 + b_cls[1];
    }
}

// ---------------------------------------------------------------------------
extern "C" void kernel_launch(void* const* d_in, const int* in_sizes, int n_in,
                              void* d_out, int out_size, void* d_ws, size_t ws_size,
                              hipStream_t stream) {
    const float* x      = (const float*)d_in[0];   // [B, T, FEAT]
    const float* W_feat = (const float*)d_in[1];   // [FUSION, FEAT]
    const float* b_feat = (const float*)d_in[2];   // [FUSION]
    const float* W_ih   = (const float*)d_in[3];   // [4H, FUSION]
    const float* b_ih   = (const float*)d_in[4];   // [4H]
    const float* W_hh   = (const float*)d_in[5];   // [4H, H]
    const float* b_hh   = (const float*)d_in[6];   // [4H]
    const float* W_cls  = (const float*)d_in[7];   // [2, H]
    const float* b_cls  = (const float*)d_in[8];   // [2]
    float* out          = (float*)d_out;           // [B, T, 2]

    // ---- workspace carve-up (bytes) ----
    char* ws = (char*)d_ws;
    const long nX   = (long)B_ * T_ * FEAT_;       // 33,554,432
    const long nWf  = (long)FUSION_ * FEAT_;       //  4,194,304
    const long nWih = (long)G4_ * FUSION_;         //  4,194,304
    const long nWhh = (long)G4_ * H_;              //  4,194,304
    const long nF   = (long)B_ * T_ * FUSION_;     //  8,388,608
    const long nBH  = (long)B_ * H_;               //    131,072

    bf16*  Xb    = (bf16*)(ws);                         ws += nX   * 2;
    bf16*  Wfb   = (bf16*)(ws);                         ws += nWf  * 2;
    bf16*  Wihb  = (bf16*)(ws);                         ws += nWih * 2;
    bf16*  Whhb  = (bf16*)(ws);                         ws += nWhh * 2;
    bf16*  F     = (bf16*)(ws);                         ws += nF   * 2;
    bf16*  h_bf0 = (bf16*)(ws);                         ws += nBH  * 2;
    bf16*  h_bf1 = (bf16*)(ws);                         ws += nBH  * 2;
    float* h_f32 = (float*)(ws);                        ws += nBH  * 4;
    float* c_st  = (float*)(ws);                        ws += nBH  * 4;
    (void)ws_size; (void)in_sizes; (void)n_in; (void)out_size;

    // 1) one-time fp32 -> bf16 conversions (x + all three big weights)
    cvt_bf16_kernel<<<2048, 256, 0, stream>>>(x,      Xb,   nX);
    cvt_bf16_kernel<<<1024, 256, 0, stream>>>(W_feat, Wfb,  nWf);
    cvt_bf16_kernel<<<1024, 256, 0, stream>>>(W_ih,   Wihb, nWih);
    cvt_bf16_kernel<<<1024, 256, 0, stream>>>(W_hh,   Whhb, nWhh);

    // 2) zero the recurrent state (h ping buffer + c)
    init_state_kernel<<<(int)((nBH + 255) / 256), 256, 0, stream>>>(h_bf0, c_st, (int)nBH);

    // 3) batched feature GEMM + ReLU for all B*T rows (hoisted out of the scan)
    //    8192/32 * 1024/32 = 8192 waves -> 1024 blocks x 8 waves
    feat_gemm_kernel<<<1024, 256, 0, stream>>>(Xb, Wfb, b_feat, F);

    // 4) sequential scan: fused (gates GEMM K=2048 + LSTM cell), then classifier
    //    512 waves/step -> 64 blocks x 8 waves; h ping-pongs in bf16
    for (int t = 0; t < T_; ++t) {
        const bf16* hin  = (t & 1) ? h_bf1 : h_bf0;
        bf16*       hout = (t & 1) ? h_bf0 : h_bf1;
        lstm_step_kernel<<<64, 256, 0, stream>>>(F, Wihb, Whhb, b_ih, b_hh,
                                                 hin, hout, h_f32, c_st, t);
        classify_kernel<<<16, 256, 0, stream>>>(h_f32, W_cls, b_cls, out, t);
    }
}
